// D3_S_atten_39831526703211
// MI455X (gfx1250) — matrix operand order repaired and checked by reading, not compiled, not run on hardware
//
#include <hip/hip_runtime.h>

typedef __attribute__((ext_vector_type(16))) _Float16 v16h;
typedef __attribute__((ext_vector_type(8)))  float    v8f;

#define NB 8
#define NC 64
#define NT 16
#define WHTOT 4096
#define KCHUNKS 18          // K = 576 = 18 * 32
#define S_OFFSET_BYTES 32768
// LDS tile: 6 rows (4 output rows + halo) x 66 cols (64 + halo) x 64 channels, f16
#define LDS_R 6
#define LDS_C 66
#define LDS_ELEMS (LDS_R * LDS_C * NC)   // 25344 halfs = 49.5 KiB

// ---------------- small setup kernels ----------------

__global__ __launch_bounds__(256) void zero_s_kernel(float* s) {
    int i = blockIdx.x * 256 + threadIdx.x;
    if (i < NB * NC * NT) s[i] = 0.f;
}

// Pack weights into per-lane WMMA A-fragment order:
// apack[((conv*4 + mt)*18 + kc)*512 + lane*16 + e]
// A 16x32 f16 order used here (contiguous K, same as B): lane<16 holds M=lane, K in {0..15}; lane>=16 holds M=lane-16, K in {16..31}
__global__ __launch_bounds__(256) void pack_w_kernel(const float* wq, const float* wk,
                                                     const float* wv, _Float16* apack) {
    int idx  = blockIdx.x * 256 + threadIdx.x;   // exactly 3*4*18*512 = 110592 threads
    int e    = idx & 15;
    int lane = (idx >> 4) & 31;
    int kc   = (idx >> 9) % KCHUNKS;
    int mt   = (idx / (512 * KCHUNKS)) & 3;
    int cv   = idx / (512 * KCHUNKS * 4);
    int lk   = lane >> 4;
    int K    = kc * 32 + e + 16 * lk;   // contiguous in-lane K order
    int tap  = K / 64;            // K ordered tap-major: k = tap*64 + ci
    int ci   = K & 63;
    int M    = mt * 16 + (lane & 15);
    const float* w = (cv == 0) ? wq : (cv == 1) ? wk : wv;  // [O=64][I=64][1][3][3]
    apack[idx] = (_Float16)w[(M * 64 + ci) * 9 + tap];
}

// ---------------- shared GEMM helpers ----------------

// Stage haloed f32 input tile -> f16 LDS, channel-innermost: xs[(r*66 + c)*64 + ci]
// 4 channels per iteration -> single ds_store_b64
__device__ __forceinline__ void fill_tile(_Float16* xs, const float* x,
                                          int b, int t, int w0, int tid) {
    const float* xbase = x + ((size_t)(b * NC) * NT + t) * WHTOT;  // + ci*(NT*WHTOT)
    for (int idx = tid; idx < LDS_R * LDS_C * (NC / 4); idx += 256) {
        int cc  = idx % LDS_C;               // consecutive threads -> consecutive h (coalesced)
        int ciq = (idx / LDS_C) & 15;
        int r   = idx / (LDS_C * (NC / 4));
        int ci0 = ciq * 4;
        int hh  = cc - 1;
        int ww  = w0 - 1 + r;
        bool ok = ((unsigned)hh < 64u) && ((unsigned)ww < 64u);
        const float* src = xbase + (size_t)ci0 * (NT * WHTOT) + ww * 64 + hh;
        _Float16 q0 = (_Float16)(ok ? src[0 * (NT * WHTOT)] : 0.f);
        _Float16 q1 = (_Float16)(ok ? src[1 * (NT * WHTOT)] : 0.f);
        _Float16 q2 = (_Float16)(ok ? src[2 * (NT * WHTOT)] : 0.f);
        _Float16 q3 = (_Float16)(ok ? src[3 * (NT * WHTOT)] : 0.f);
        typedef __attribute__((ext_vector_type(4))) _Float16 v4h;
        v4h q = {q0, q1, q2, q3};
        *(v4h*)(xs + (r * LDS_C + cc) * NC + ci0) = q;
    }
}

// Per-chunk scalar LDS delta: tap -> (dy*66+dx)*64 plus ci sub-block select
__device__ __forceinline__ int chunk_delta(int kc, int lk) {
    int tap = kc >> 1;
    int dy  = tap / 3;
    int dx  = tap % 3;
    int ci0 = ((kc & 1) << 5) + (lk << 4);
    return (dy * LDS_C + dx) * NC + ci0;
}

// ---------------- Phase A: k,v GEMMs -> s[b,c,t] += sum_p k*v ----------------

__global__ __launch_bounds__(256) void qkv_phaseA(const float* __restrict__ x,
                                                  const _Float16* __restrict__ apack,
                                                  const float* __restrict__ bk,
                                                  const float* __restrict__ bv,
                                                  float* __restrict__ s) {
    __shared__ __align__(32) _Float16 xs[LDS_ELEMS];
    int tid = threadIdx.x;
    int rt = blockIdx.x, t = blockIdx.y, b = blockIdx.z;
    int w0 = rt * 4;
    fill_tile(xs, x, b, t, w0, tid);
    __syncthreads();

    int lane = tid & 31, wave = tid >> 5;
    int mt = wave & 3;        // M-tile (16 output channels)
    int nh = wave >> 2;       // which 128-pixel half of the 256-pixel tile
    int lk = lane >> 4, ln = lane & 15;

    // per-lane invariant LDS base offsets for the 8 N-subtiles
    int base[8];
#pragma unroll
    for (int n = 0; n < 8; ++n) {
        int pl = nh * 128 + n * 16 + ln;   // local pixel in 256-pixel tile
        int pw = pl >> 6, ph = pl & 63;
        base[n] = (pw * LDS_C + ph) * NC;
    }

    v8f z = {0.f, 0.f, 0.f, 0.f, 0.f, 0.f, 0.f, 0.f};
    v8f kacc[8] = {z, z, z, z, z, z, z, z};
    v8f vacc[8] = {z, z, z, z, z, z, z, z};

    const _Float16* akp = apack + ((size_t)((1 * 4 + mt) * KCHUNKS)) * 512 + lane * 16;
    const _Float16* avp = apack + ((size_t)((2 * 4 + mt) * KCHUNKS)) * 512 + lane * 16;

    for (int kc = 0; kc < KCHUNKS; ++kc) {
        v16h ak = *(const v16h*)(akp + (size_t)kc * 512);
        v16h av = *(const v16h*)(avp + (size_t)kc * 512);
        int d = chunk_delta(kc, lk);
#pragma unroll
        for (int n = 0; n < 8; ++n) {
            v16h bfrag = *(const v16h*)(xs + base[n] + d);
            kacc[n] = __builtin_amdgcn_wmma_f32_16x16x32_f16(false, ak, false, bfrag,
                                                             (short)0, kacc[n], false, false);
            vacc[n] = __builtin_amdgcn_wmma_f32_16x16x32_f16(false, av, false, bfrag,
                                                             (short)0, vacc[n], false, false);
        }
    }

    // (k + bk)*(v + bv): sum over 8 subtiles in registers, then over 16 pixels per half-wave
#pragma unroll
    for (int r = 0; r < 8; ++r) {
        int ch = mt * 16 + r + 8 * lk;
        float bkv = bk[ch], bvv = bv[ch];
        float kv = 0.f;
#pragma unroll
        for (int n = 0; n < 8; ++n)
            kv += (kacc[n][r] + bkv) * (vacc[n][r] + bvv);
        kv += __shfl_xor(kv, 1, 32);
        kv += __shfl_xor(kv, 2, 32);
        kv += __shfl_xor(kv, 4, 32);
        kv += __shfl_xor(kv, 8, 32);
        if (ln == 0) atomicAdd(&s[(b * NC + ch) * NT + t], kv);
    }
}

// ---------------- Phase B: q GEMM, out = gamma*q*s + x ----------------

__global__ __launch_bounds__(256) void qs_phaseB(const float* __restrict__ x,
                                                 const _Float16* __restrict__ apack,
                                                 const float* __restrict__ s,
                                                 const float* __restrict__ bq,
                                                 const float* __restrict__ gamma,
                                                 float* __restrict__ out) {
    __shared__ __align__(32) _Float16 xs[LDS_ELEMS];
    int tid = threadIdx.x;
    int rt = blockIdx.x, t = blockIdx.y, b = blockIdx.z;
    int w0 = rt * 4;
    fill_tile(xs, x, b, t, w0, tid);
    __syncthreads();

    int lane = tid & 31, wave = tid >> 5;
    int mt = wave & 3;
    int nh = wave >> 2;
    int lk = lane >> 4, ln = lane & 15;

    int base[8];
#pragma unroll
    for (int n = 0; n < 8; ++n) {
        int pl = nh * 128 + n * 16 + ln;
        int pw = pl >> 6, ph = pl & 63;
        base[n] = (pw * LDS_C + ph) * NC;
    }

    v8f z = {0.f, 0.f, 0.f, 0.f, 0.f, 0.f, 0.f, 0.f};
    v8f qacc[8] = {z, z, z, z, z, z, z, z};

    const _Float16* aqp = apack + ((size_t)((0 * 4 + mt) * KCHUNKS)) * 512 + lane * 16;

    for (int kc = 0; kc < KCHUNKS; ++kc) {
        v16h aq = *(const v16h*)(aqp + (size_t)kc * 512);
        int d = chunk_delta(kc, lk);
#pragma unroll
        for (int n = 0; n < 8; ++n) {
            v16h bfrag = *(const v16h*)(xs + base[n] + d);
            qacc[n] = __builtin_amdgcn_wmma_f32_16x16x32_f16(false, aq, false, bfrag,
                                                             (short)0, qacc[n], false, false);
        }
    }

    float g = gamma[0];
#pragma unroll
    for (int r = 0; r < 8; ++r) {
        int ch = mt * 16 + r + 8 * lk;
        float qb = bq[ch];
        float sv = s[(b * NC + ch) * NT + t];
        size_t chbase = (((size_t)(b * NC + ch) * NT) + t) * WHTOT;
#pragma unroll
        for (int n = 0; n < 8; ++n) {
            int p = rt * 256 + nh * 128 + n * 16 + ln;
            size_t oi = chbase + p;
            out[oi] = g * (qacc[n][r] + qb) * sv + x[oi];
        }
    }
}

// ---------------- launch ----------------

extern "C" void kernel_launch(void* const* d_in, const int* in_sizes, int n_in,
                              void* d_out, int out_size, void* d_ws, size_t ws_size,
                              hipStream_t stream) {
    const float* x     = (const float*)d_in[0];
    const float* wq    = (const float*)d_in[1];
    const float* wk    = (const float*)d_in[2];
    const float* wv    = (const float*)d_in[3];
    const float* bq    = (const float*)d_in[4];
    const float* bk    = (const float*)d_in[5];
    const float* bv    = (const float*)d_in[6];
    const float* gamma = (const float*)d_in[7];
    float* out = (float*)d_out;

    float*    s     = (float*)d_ws;                                 // 8192 floats
    _Float16* apack = (_Float16*)((char*)d_ws + S_OFFSET_BYTES);    // 110592 halfs

    zero_s_kernel<<<32, 256, 0, stream>>>(s);
    pack_w_kernel<<<432, 256, 0, stream>>>(wq, wk, wv, apack);      // 432*256 == 110592

    dim3 grid(16, NT, NB);   // 16 four-row tiles x 16 time x 8 batch
    qkv_phaseA<<<grid, 256, 0, stream>>>(x, apack, bk, bv, s);
    qs_phaseB<<<grid, 256, 0, stream>>>(x, apack, s, bq, gamma, out);
}
